// CATLayer_75651553951840
// MI455X (gfx1250) — matrix operand order repaired and checked by reading, not compiled, not run on hardware
//
#include <hip/hip_runtime.h>

// ---------------------------------------------------------------------------
// CDNA5 (gfx1250) fused MoE-FFN + cross-expert attention layer.
// All dense math: v_wmma_f32_16x16x32_bf16 (wave32), 128x128x128 block tiles,
// double-buffered LDS fed by global_load_async_to_lds_b128 (ASYNCcnt) when
// the toolchain exposes it (guarded fallback: plain loads).
// Weights are pre-converted fp32->bf16 once so every GEMM operand is bf16.
// Only q / out-proj for the selected expert row are computed (8x saving);
// expert id is read on-device (graph-capture safe).
// ---------------------------------------------------------------------------

typedef __attribute__((ext_vector_type(16))) __bf16 v16bf;
typedef __attribute__((ext_vector_type(8)))  __bf16 v8bf;
typedef __attribute__((ext_vector_type(8)))  float  v8f;
typedef __attribute__((ext_vector_type(4)))  int    v4i;

#define KC 128
#define LDS_STRIDE 136  // 128 + 8 halfs pad -> 272B row stride (16B aligned)

#if defined(__AMDGCN__) && \
    __has_builtin(__builtin_amdgcn_global_load_async_to_lds_b128) && \
    __has_builtin(__builtin_amdgcn_s_wait_asynccnt)
#define ASYNC_COPY 1
typedef __attribute__((address_space(1))) v4i gv4i;  // global 128-bit vec
typedef __attribute__((address_space(3))) v4i lv4i;  // LDS    128-bit vec
#else
#define ASYNC_COPY 0
#endif

__device__ inline unsigned pack2bf(float a, float b) {
  __bf16 x = (__bf16)a, y = (__bf16)b;
  unsigned short sx, sy;
  __builtin_memcpy(&sx, &x, 2);
  __builtin_memcpy(&sy, &y, 2);
  return (unsigned)sx | ((unsigned)sy << 16);
}

// 16B global -> LDS copy: async DMA on CDNA5, sync fallback otherwise.
__device__ inline void copy16(const __bf16* __restrict__ g, __bf16* l) {
#if ASYNC_COPY
  __builtin_amdgcn_global_load_async_to_lds_b128((gv4i*)g, (lv4i*)l, 0, 0);
#else
  *(uint4*)l = *(const uint4*)g;
#endif
}

template <int N>
__device__ inline void wait_async() {
#if ASYNC_COPY
  __builtin_amdgcn_s_wait_asynccnt(N);
#endif
}

// ---- WMMA fragment gathers from LDS (bf16 16x16x32 layouts, ISA 7.12.2) ---
// A 16x32 (MxK): lanes 0-15 -> M=lane, K {0..7,16..23}; lanes 16-31 -> K {8..15,24..31}
__device__ inline v16bf loadAfrag(const __bf16* T, int rowBase, int k0, int lane) {
  int l = lane & 15, hi = lane >> 4;
  const __bf16* p = T + (rowBase + l) * LDS_STRIDE + k0 + hi * 8;
  v8bf lo = *(const v8bf*)p;
  v8bf hv = *(const v8bf*)(p + 16);
  return __builtin_shufflevector(lo, hv, 0,1,2,3,4,5,6,7,8,9,10,11,12,13,14,15);
}
// B 32x16 (KxN): lanes 0-15 -> N=lane, K {0..7,16..23}; lanes 16-31 -> N=lane-16, K {8..15,24..31}
__device__ inline v16bf loadBfrag(const __bf16* T, int rowBase, int k0, int lane) {
  int l = lane & 15, hi = lane >> 4;
  const __bf16* p = T + (rowBase + l) * LDS_STRIDE + k0 + hi * 8;
  v8bf lo = *(const v8bf*)p;
  v8bf hv = *(const v8bf*)(p + 16);
  return __builtin_shufflevector(lo, hv, 0,1,2,3,4,5,6,7,8,9,10,11,12,13,14,15);
}

// ---------------------------------------------------------------------------
// fp32 -> bf16 bulk conversion (8 elems / thread, RNE)
// ---------------------------------------------------------------------------
__global__ __launch_bounds__(256)
void cvt_f32_bf16_kernel(const float* __restrict__ s, __bf16* __restrict__ d, long n) {
  long i = ((long)blockIdx.x * 256 + threadIdx.x) * 8;
  if (i >= n) return;
  float4 f0 = *(const float4*)(s + i);
  float4 f1 = *(const float4*)(s + i + 4);
  uint4 u;
  u.x = pack2bf(f0.x, f0.y);
  u.y = pack2bf(f0.z, f0.w);
  u.z = pack2bf(f1.x, f1.y);
  u.w = pack2bf(f1.z, f1.w);
  *(uint4*)(d + i) = u;
}

// ---------------------------------------------------------------------------
// C[M,N] = act(A[M,K] * W[N,K]^T + bias[N]), row-major, dims % 128 == 0.
// Block: 256 thr = 8 waves, 128x128 C tile; wave = 32x64 (8 accumulators).
// K staged in 128-wide LDS tiles, double-buffered, async-copied.
// Optional device-side A base offset: A += (*aIdx) * aOffScale.
// ---------------------------------------------------------------------------
template <bool RELU, bool OUT_BF16>
__global__ __launch_bounds__(256)
void gemm_bias_kernel(const __bf16* __restrict__ A, long lda, long aOffScale,
                      const int* __restrict__ aIdx,
                      const __bf16* __restrict__ W, long ldw,
                      const float* __restrict__ bias,
                      void* __restrict__ Cout, long ldc, int K) {
  __shared__ __align__(16) __bf16 As[2][128 * LDS_STRIDE];
  __shared__ __align__(16) __bf16 Bs[2][128 * LDS_STRIDE];

  const int tid  = threadIdx.x;
  const int lane = tid & 31, wid = tid >> 5;
  const int mBlock = blockIdx.y * 128, nBlock = blockIdx.x * 128;
  const int mWave  = (wid & 3) * 32,  nWave  = (wid >> 2) * 64;

  if (aIdx) A += (long)(*aIdx) * aOffScale;

  v8f acc[2][4] = {};

  // copy slice: 16 col-groups of 8 halfs x 8 row steps per matrix
  const int cg = tid & 15;  // 0..15
  const int r0 = tid >> 4;  // 0..15

  auto issue_tile = [&](int buf, int kt) {
#pragma unroll
    for (int j = 0; j < 8; ++j) {
      int r = r0 + j * 16;
      copy16(A + (long)(mBlock + r) * lda + kt + cg * 8,
             &As[buf][r * LDS_STRIDE + cg * 8]);
      copy16(W + (long)(nBlock + r) * ldw + kt + cg * 8,
             &Bs[buf][r * LDS_STRIDE + cg * 8]);
    }
  };

  issue_tile(0, 0);
  int cur = 0;

  for (int kt = 0; kt < K; kt += KC) {
    if (kt + KC < K) {
      issue_tile(cur ^ 1, kt + KC);  // stream next tile while computing
      wait_async<16>();              // current tile's 16 async ops done
    } else {
      wait_async<0>();
    }
    __syncthreads();

    const __bf16* Ac = As[cur];
    const __bf16* Bc = Bs[cur];
#pragma unroll
    for (int ks = 0; ks < KC; ks += 32) {
      v16bf a0 = loadAfrag(Ac, mWave + 0,  ks, lane);
      v16bf a1 = loadAfrag(Ac, mWave + 16, ks, lane);
      v16bf b0 = loadBfrag(Bc, nWave + 0,  ks, lane);
      v16bf b1 = loadBfrag(Bc, nWave + 16, ks, lane);
      v16bf b2 = loadBfrag(Bc, nWave + 32, ks, lane);
      v16bf b3 = loadBfrag(Bc, nWave + 48, ks, lane);
      acc[0][0] = __builtin_amdgcn_wmma_f32_16x16x32_bf16(false, a0, false, b0, (short)0, acc[0][0], false, false);
      acc[1][0] = __builtin_amdgcn_wmma_f32_16x16x32_bf16(false, a1, false, b0, (short)0, acc[1][0], false, false);
      acc[0][1] = __builtin_amdgcn_wmma_f32_16x16x32_bf16(false, a0, false, b1, (short)0, acc[0][1], false, false);
      acc[1][1] = __builtin_amdgcn_wmma_f32_16x16x32_bf16(false, a1, false, b1, (short)0, acc[1][1], false, false);
      acc[0][2] = __builtin_amdgcn_wmma_f32_16x16x32_bf16(false, a0, false, b2, (short)0, acc[0][2], false, false);
      acc[1][2] = __builtin_amdgcn_wmma_f32_16x16x32_bf16(false, a1, false, b2, (short)0, acc[1][2], false, false);
      acc[0][3] = __builtin_amdgcn_wmma_f32_16x16x32_bf16(false, a0, false, b3, (short)0, acc[0][3], false, false);
      acc[1][3] = __builtin_amdgcn_wmma_f32_16x16x32_bf16(false, a1, false, b3, (short)0, acc[1][3], false, false);
    }
    __syncthreads();  // all waves done reading buf before it is refilled
    cur ^= 1;
  }

  // C/D layout: VGPR i -> row = i + (lane>=16)*8 ; col = lane&15
  const int l = lane & 15, hi = lane >> 4;
#pragma unroll
  for (int j = 0; j < 4; ++j) {
    int col = nBlock + nWave + j * 16 + l;
    float bj = bias[col];
#pragma unroll
    for (int i = 0; i < 2; ++i) {
#pragma unroll
      for (int r = 0; r < 8; ++r) {
        int row = mBlock + mWave + i * 16 + hi * 8 + r;
        float v = acc[i][j][r] + bj;
        if (RELU) v = fmaxf(v, 0.0f);
        if (OUT_BF16)
          ((__bf16*)Cout)[(long)row * ldc + col] = (__bf16)v;
        else
          ((float*)Cout)[(long)row * ldc + col] = v;
      }
    }
  }
}

// ---------------------------------------------------------------------------
// Cross-expert attention: one wave32 per (token, head). NE=8 keys, DH=64.
// logits = q.k/sqrt(64) + tril-mask row (k<=eid ? 1 : 0); softmax; o = p.v
// ---------------------------------------------------------------------------
__global__ __launch_bounds__(256)
void attn_kernel(const __bf16* __restrict__ q3, const __bf16* __restrict__ kall,
                 const __bf16* __restrict__ vall, const int* __restrict__ expertPtr,
                 __bf16* __restrict__ o3) {
  const int NE = 8, H = 16, DH = 64, E = 1024;
  int tid = threadIdx.x, lane = tid & 31, wid = tid >> 5;
  int gw = blockIdx.x * 8 + wid;
  int t = gw / H, h = gw % H;
  int eid = *expertPtr;
  int d0 = lane * 2;  // 2 head-dims per lane (32 lanes * 2 = DH)

  const __bf16* qp = q3 + (long)t * E + h * DH + d0;
  float q0 = (float)qp[0], q1 = (float)qp[1];

  float logits[8];
#pragma unroll
  for (int n = 0; n < NE; ++n) {
    const __bf16* kp = kall + ((long)t * NE + n) * E + h * DH + d0;
    float part = q0 * (float)kp[0] + q1 * (float)kp[1];
#pragma unroll
    for (int m = 16; m >= 1; m >>= 1) part += __shfl_xor(part, m, 32);
    logits[n] = part * 0.125f + ((n <= eid) ? 1.0f : 0.0f);
  }
  float mx = logits[0];
#pragma unroll
  for (int n = 1; n < NE; ++n) mx = fmaxf(mx, logits[n]);
  float p[8], s = 0.0f;
#pragma unroll
  for (int n = 0; n < NE; ++n) { p[n] = __expf(logits[n] - mx); s += p[n]; }
  float inv = 1.0f / s;
  float o0 = 0.0f, o1 = 0.0f;
#pragma unroll
  for (int n = 0; n < NE; ++n) {
    const __bf16* vp = vall + ((long)t * NE + n) * E + h * DH + d0;
    float w = p[n] * inv;
    o0 += w * (float)vp[0];
    o1 += w * (float)vp[1];
  }
  __bf16* op = o3 + (long)t * E + h * DH + d0;
  op[0] = (__bf16)o0;
  op[1] = (__bf16)o1;
}

// ---------------------------------------------------------------------------
extern "C" void kernel_launch(void* const* d_in, const int* in_sizes, int n_in,
                              void* d_out, int out_size, void* d_ws, size_t ws_size,
                              hipStream_t stream) {
  (void)in_sizes; (void)n_in; (void)out_size; (void)ws_size;
  const int B = 4, S = 2048, NE = 8, E = 1024, H = 16;
  const int M = B * S;    // 8192 tokens
  const int F = 4 * E;    // 4096 hidden

  const float* x  = (const float*)d_in[0];
  const float* w1 = (const float*)d_in[1];
  const float* b1 = (const float*)d_in[2];
  const float* w2 = (const float*)d_in[3];
  const float* b2 = (const float*)d_in[4];
  const float* pw = (const float*)d_in[5];  // in_proj_w [3E, E]
  const float* pb = (const float*)d_in[6];
  const float* ow = (const float*)d_in[7];
  const float* ob = (const float*)d_in[8];
  const int*  eid = (const int*)d_in[9];

  char* ws = (char*)d_ws;
  __bf16* xb   = (__bf16*)ws; ws += (size_t)M * E * 2;            //  16 MB
  __bf16* w1b  = (__bf16*)ws; ws += (size_t)NE * F * E * 2;       //  64 MB
  __bf16* w2b  = (__bf16*)ws; ws += (size_t)NE * E * F * 2;       //  64 MB
  __bf16* pwb  = (__bf16*)ws; ws += (size_t)3 * E * E * 2;        //   6 MB
  __bf16* owb  = (__bf16*)ws; ws += (size_t)E * E * 2;            //   2 MB
  __bf16* h    = (__bf16*)ws; ws += (size_t)M * F * 2;            //  64 MB
  __bf16* eo   = (__bf16*)ws; ws += (size_t)M * NE * E * 2;       // 128 MB
  __bf16* kall = (__bf16*)ws; ws += (size_t)M * NE * E * 2;       // 128 MB
  __bf16* vall = (__bf16*)ws; ws += (size_t)M * NE * E * 2;       // 128 MB
  __bf16* q3   = (__bf16*)ws; ws += (size_t)M * E * 2;            //  16 MB
  __bf16* o3   = (__bf16*)ws; ws += (size_t)M * E * 2;            //  16 MB

  dim3 blk(256);

  // ---- one-shot fp32 -> bf16 conversions ---------------------------------
  {
    long nx = (long)M * E, n1 = (long)NE * F * E, n2 = (long)NE * E * F;
    long np = (long)3 * E * E, no = (long)E * E;
    cvt_f32_bf16_kernel<<<dim3(nx / (8 * 256)), blk, 0, stream>>>(x,  xb,  nx);
    cvt_f32_bf16_kernel<<<dim3(n1 / (8 * 256)), blk, 0, stream>>>(w1, w1b, n1);
    cvt_f32_bf16_kernel<<<dim3(n2 / (8 * 256)), blk, 0, stream>>>(w2, w2b, n2);
    cvt_f32_bf16_kernel<<<dim3(np / (8 * 256)), blk, 0, stream>>>(pw, pwb, np);
    cvt_f32_bf16_kernel<<<dim3(no / (8 * 256)), blk, 0, stream>>>(ow, owb, no);
  }

  // ---- MoE FFN: h = relu(x w1^T + b1); eo[:,n,:] = h w2^T + b2 ------------
  for (int n = 0; n < NE; ++n) {
    gemm_bias_kernel<true, true>
        <<<dim3(F / 128, M / 128), blk, 0, stream>>>(
            xb, E, 0, nullptr, w1b + (size_t)n * F * E, E, b1 + (size_t)n * F,
            h, F, E);
    gemm_bias_kernel<false, true>
        <<<dim3(E / 128, M / 128), blk, 0, stream>>>(
            h, F, 0, nullptr, w2b + (size_t)n * E * F, F, b2 + (size_t)n * E,
            eo + (size_t)n * E, (long)NE * E, F);
  }

  // ---- k, v for all (token, expert) rows: [M*NE, E] -----------------------
  gemm_bias_kernel<false, true>
      <<<dim3(E / 128, (M * NE) / 128), blk, 0, stream>>>(
          eo, E, 0, nullptr, pwb + (size_t)E * E, E, pb + E, kall, E, E);
  gemm_bias_kernel<false, true>
      <<<dim3(E / 128, (M * NE) / 128), blk, 0, stream>>>(
          eo, E, 0, nullptr, pwb + (size_t)2 * E * E, E, pb + 2 * E, vall, E, E);

  // ---- q only for the selected expert row (device-side index) ------------
  gemm_bias_kernel<false, true>
      <<<dim3(E / 128, M / 128), blk, 0, stream>>>(
          eo, (long)NE * E, E, eid, pwb, E, pb, q3, E, E);

  // ---- softmax attention across experts -----------------------------------
  attn_kernel<<<dim3((M * H) / 8), blk, 0, stream>>>(q3, kall, vall, eid, o3);

  // ---- output projection -> fp32 d_out ------------------------------------
  gemm_bias_kernel<false, false>
      <<<dim3(E / 128, M / 128), blk, 0, stream>>>(
          o3, E, 0, nullptr, owb, E, ob, d_out, E, E);
}
